// SelfAttentionModule_65481071402447
// MI455X (gfx1250) — matrix operand order repaired and checked by reading, not compiled, not run on hardware
//
#include <hip/hip_runtime.h>
#include <hip/hip_bf16.h>

#define NN 8192
#define DD 1024

typedef __attribute__((ext_vector_type(16))) _Float16 v16h;
typedef __attribute__((ext_vector_type(8)))  _Float16 v8h;
typedef __attribute__((ext_vector_type(8)))  float    v8f;

static __device__ __forceinline__ v16h cat16(v8h lo, v8h hi) {
  return __builtin_shufflevector(lo, hi, 0,1,2,3,4,5,6,7,8,9,10,11,12,13,14,15);
}

// A fragment (16x32 f16, lane = M row): lane<16 holds K 0..15,
// lane>=16 holds K 16..31 (matches B fragment order).
static __device__ __forceinline__ v16h ldA16(const _Float16* p, int ld, int row,
                                             int k0, int lane) {
  const _Float16* q = p + (size_t)row * ld + k0 + ((lane & 16) ? 16 : 0);
  v8h lo = *(const v8h*)q;
  v8h hi = *(const v8h*)(q + 8);
  return cat16(lo, hi);
}

// B fragment (32x16 f16, lane = N col): lane<16 holds K 0..15, lane>=16 K 16..31.
static __device__ __forceinline__ v16h ldB16(const _Float16* p, int ld, int col,
                                             int k0, int lane) {
  const _Float16* q = p + (size_t)col * ld + k0 + ((lane & 16) ? 16 : 0);
  v8h lo = *(const v8h*)q;
  v8h hi = *(const v8h*)(q + 8);
  return cat16(lo, hi);
}

static __device__ __forceinline__ v8f wmma_f16(v16h a, v16h b, v8f c) {
  return __builtin_amdgcn_wmma_f32_16x16x32_f16(false, a, false, b, (short)0, c,
                                                false, false);
}

// gfx1250 async DMA: global -> LDS, 16B per lane, tracked by ASYNCcnt.
static __device__ __forceinline__ void async_b128(unsigned lds_off, const void* g) {
  asm volatile("global_load_async_to_lds_b128 %0, %1, off"
               :: "v"(lds_off), "v"(g) : "memory");
}
static __device__ __forceinline__ void wait_async0() {
  asm volatile("s_wait_asynccnt 0x0" ::: "memory");
}
// Partial wait: retire only the oldest in-flight stage (async loads complete
// in order, ISA 08 §4.1) so the next panel keeps streaming during compute.
static __device__ __forceinline__ void wait_async4() {
  asm volatile("s_wait_asynccnt 0x4" ::: "memory");
}

// ---------------------------------------------------------------------------
// Prep: f32->f16 convert + transpose; also K = X*diag(e) and plain Xh.
// ---------------------------------------------------------------------------
__global__ void prep_transpose(const float* __restrict__ src, int rows, int cols,
                               _Float16* __restrict__ dstT,   // [cols x rows]
                               const float* __restrict__ ent,
                               _Float16* __restrict__ dstK,   // [rows x cols]
                               _Float16* __restrict__ dstX) { // [rows x cols]
  __shared__ float tile[16][17];
  int tx = threadIdx.x, ty = threadIdx.y;
  int c = blockIdx.x * 16 + tx;
  int r = blockIdx.y * 16 + ty;
  float v = src[(size_t)r * cols + c];
  if (dstK) dstK[(size_t)r * cols + c] = (_Float16)(v * ent[c]);
  if (dstX) dstX[(size_t)r * cols + c] = (_Float16)v;
  tile[ty][tx] = v;
  __syncthreads();
  int c2 = blockIdx.x * 16 + ty;
  int r2 = blockIdx.y * 16 + tx;
  dstT[(size_t)c2 * rows + r2] = (_Float16)tile[tx][ty];
}

// ---------------------------------------------------------------------------
// Q = (X @ R) / sqrt(D) -> f16.  A from Xh (f16), B from Rt.  64x64 per wave.
// ---------------------------------------------------------------------------
__global__ __launch_bounds__(128) void q_gemm(const _Float16* __restrict__ Xh,
                                              const _Float16* __restrict__ Rt,
                                              _Float16* __restrict__ Qh) {
  int lane = threadIdx.x & 31, w = threadIdx.x >> 5;
  int m0 = blockIdx.y * 128 + (w >> 1) * 64;
  int n0 = blockIdx.x * 128 + (w & 1) * 64;
  v8f acc[4][4] = {};
  for (int k0 = 0; k0 < DD; k0 += 32) {
    if (k0 + 64 < DD) {
      __builtin_prefetch(Xh + (size_t)m0 * DD + k0 + 64, 0, 1);
      __builtin_prefetch(Rt + (size_t)n0 * DD + k0 + 64, 0, 1);
    }
    v16h a[4], b[4];
#pragma unroll
    for (int i = 0; i < 4; i++) a[i] = ldA16(Xh, DD, m0 + i * 16, k0, lane);
#pragma unroll
    for (int j = 0; j < 4; j++) b[j] = ldB16(Rt, DD, n0 + j * 16, k0, lane);
#pragma unroll
    for (int i = 0; i < 4; i++)
#pragma unroll
      for (int j = 0; j < 4; j++) acc[i][j] = wmma_f16(a[i], b[j], acc[i][j]);
  }
  const float sc = 0.03125f;  // 1/sqrt(1024)
  int cr = (lane & 16) ? 8 : 0, cc = lane & 15;
#pragma unroll
  for (int i = 0; i < 4; i++)
    for (int j = 0; j < 4; j++)
#pragma unroll
      for (int r = 0; r < 8; r++)
        Qh[(size_t)(m0 + i * 16 + r + cr) * DD + (n0 + j * 16 + cc)] =
            (_Float16)(acc[i][j][r] * sc);
}

// ---------------------------------------------------------------------------
// S = Q @ K^T (raw logits, f16).  K panel double-buffered in LDS via async
// DMA (ASYNCcnt ping-pong); B fragments served from LDS, shared by 4 waves.
// ---------------------------------------------------------------------------
__global__ __launch_bounds__(128) void qk_gemm(const _Float16* __restrict__ Q,
                                               const _Float16* __restrict__ K,
                                               _Float16* __restrict__ S) {
  __shared__ alignas(16) _Float16 sK[2][128 * 32];  // 2 x 8 KB panels
  int t = threadIdx.x;
  int lane = t & 31, w = t >> 5;
  int m0 = blockIdx.y * 128 + (w >> 1) * 64;  // wave's query rows
  int nb = blockIdx.x * 128;                  // block's key base
  int nw = (w & 1) * 64;                      // wave's key offset in panel
  const _Float16* krow = K + (size_t)(nb + t) * DD;  // thread's key row
  unsigned ldsrow0 = (unsigned)(size_t)(void*)&sK[0][t * 32];
  unsigned ldsrow1 = (unsigned)(size_t)(void*)&sK[1][t * 32];

  // Prologue: stages 0 and 1 in flight (4 async b128 per stage per wave).
#pragma unroll
  for (int c = 0; c < 4; c++) async_b128(ldsrow0 + c * 16, krow + 0 + c * 8);
#pragma unroll
  for (int c = 0; c < 4; c++) async_b128(ldsrow1 + c * 16, krow + 32 + c * 8);

  v8f acc[4][4] = {};
  for (int k0 = 0; k0 < DD; k0 += 32) {
    int cur = (k0 >> 5) & 1;
    // Retire the oldest stage (this iteration's panel); keep next streaming.
    if (k0 + 32 < DD) wait_async4(); else wait_async0();
    __syncthreads();  // panel `cur` complete in all waves

    v16h a[4], b[4];
#pragma unroll
    for (int i = 0; i < 4; i++) a[i] = ldA16(Q, DD, m0 + i * 16, k0, lane);
    const _Float16* panel = sK[cur];
#pragma unroll
    for (int j = 0; j < 4; j++) {
      const _Float16* q =
          &panel[(nw + j * 16 + (lane & 15)) * 32 + ((lane & 16) ? 16 : 0)];
      b[j] = cat16(*(const v8h*)q, *(const v8h*)(q + 8));
    }
#pragma unroll
    for (int i = 0; i < 4; i++)
#pragma unroll
      for (int j = 0; j < 4; j++) acc[i][j] = wmma_f16(a[i], b[j], acc[i][j]);

    __syncthreads();  // all waves done reading `cur` before we overwrite it
    if (k0 + 64 < DD) {
      unsigned dst = cur ? ldsrow1 : ldsrow0;
#pragma unroll
      for (int c = 0; c < 4; c++)
        async_b128(dst + c * 16, krow + k0 + 64 + c * 8);
    }
  }

  int cr = (lane & 16) ? 8 : 0, cc = lane & 15;
#pragma unroll
  for (int i = 0; i < 4; i++)
    for (int j = 0; j < 4; j++)
#pragma unroll
      for (int r = 0; r < 8; r++)
        S[(size_t)(m0 + i * 16 + r + cr) * NN + (nb + nw + j * 16 + cc)] =
            (_Float16)acc[i][j][r];
}

// ---------------------------------------------------------------------------
// In-place row softmax over 8192-wide rows (one block per row, L2-hot).
// ---------------------------------------------------------------------------
__global__ void row_softmax(_Float16* __restrict__ S) {
  __shared__ float red[256];
  _Float16* row = S + (size_t)blockIdx.x * NN;
  int t = threadIdx.x;
  float m = -1e30f;
  for (int i = t; i < NN; i += 256) m = fmaxf(m, (float)row[i]);
  red[t] = m;
  __syncthreads();
  for (int s = 128; s > 0; s >>= 1) {
    if (t < s) red[t] = fmaxf(red[t], red[t + s]);
    __syncthreads();
  }
  m = red[0];
  __syncthreads();
  float sum = 0.f;
  for (int i = t; i < NN; i += 256) sum += __expf((float)row[i] - m);
  red[t] = sum;
  __syncthreads();
  for (int s = 128; s > 0; s >>= 1) {
    if (t < s) red[t] += red[t + s];
    __syncthreads();
  }
  float inv = 1.0f / red[0];
  for (int i = t; i < NN; i += 256)
    row[i] = (_Float16)(__expf((float)row[i] - m) * inv);
}

// ---------------------------------------------------------------------------
// out = P @ X (f32).  B operand from Vt (X transposed, key-contiguous).
// ---------------------------------------------------------------------------
__global__ __launch_bounds__(128) void pv_gemm(const _Float16* __restrict__ P,
                                               const _Float16* __restrict__ Vt,
                                               float* __restrict__ O) {
  int lane = threadIdx.x & 31, w = threadIdx.x >> 5;
  int m0 = blockIdx.y * 128 + (w >> 1) * 64;
  int n0 = blockIdx.x * 128 + (w & 1) * 64;
  v8f acc[4][4] = {};
  for (int k0 = 0; k0 < NN; k0 += 32) {
    if (k0 + 64 < NN) {
      __builtin_prefetch(P + (size_t)m0 * NN + k0 + 64, 0, 1);
      __builtin_prefetch(Vt + (size_t)n0 * NN + k0 + 64, 0, 1);
    }
    v16h a[4], b[4];
#pragma unroll
    for (int i = 0; i < 4; i++) a[i] = ldA16(P, NN, m0 + i * 16, k0, lane);
#pragma unroll
    for (int j = 0; j < 4; j++) b[j] = ldB16(Vt, NN, n0 + j * 16, k0, lane);
#pragma unroll
    for (int i = 0; i < 4; i++)
#pragma unroll
      for (int j = 0; j < 4; j++) acc[i][j] = wmma_f16(a[i], b[j], acc[i][j]);
  }
  int cr = (lane & 16) ? 8 : 0, cc = lane & 15;
#pragma unroll
  for (int i = 0; i < 4; i++)
    for (int j = 0; j < 4; j++)
#pragma unroll
      for (int r = 0; r < 8; r++)
        O[(size_t)(m0 + i * 16 + r + cr) * DD + (n0 + j * 16 + cc)] =
            acc[i][j][r];
}

extern "C" void kernel_launch(void* const* d_in, const int* in_sizes, int n_in,
                              void* d_out, int out_size, void* d_ws, size_t ws_size,
                              hipStream_t stream) {
  const float* X = (const float*)d_in[0];  // [8192,1024]
  const float* R = (const float*)d_in[1];  // [1024,1024]
  const float* E = (const float*)d_in[2];  // [1024]
  float* out = (float*)d_out;              // [8192,1024]

  char* ws = (char*)d_ws;
  _Float16* Qh = (_Float16*)(ws);                       // 16 MB
  _Float16* Kh = (_Float16*)(ws + ((size_t)16 << 20));  // 16 MB
  _Float16* Vt = (_Float16*)(ws + ((size_t)32 << 20));  // 16 MB
  _Float16* Rt = (_Float16*)(ws + ((size_t)48 << 20));  //  2 MB
  _Float16* Xh = (_Float16*)(ws + ((size_t)50 << 20));  // 16 MB
  _Float16* S  = (_Float16*)(ws + ((size_t)66 << 20));  // 128 MB

  dim3 b16(16, 16);
  prep_transpose<<<dim3(DD / 16, NN / 16), b16, 0, stream>>>(X, NN, DD, Vt, E, Kh,
                                                             Xh);
  prep_transpose<<<dim3(DD / 16, DD / 16), b16, 0, stream>>>(R, DD, DD, Rt, nullptr,
                                                             nullptr, nullptr);
  q_gemm<<<dim3(DD / 128, NN / 128), 128, 0, stream>>>(Xh, Rt, Qh);
  qk_gemm<<<dim3(NN / 128, NN / 128), 128, 0, stream>>>(Qh, Kh, S);
  row_softmax<<<NN, 256, 0, stream>>>(S);
  pv_gemm<<<dim3(DD / 128, NN / 128), 128, 0, stream>>>(S, Vt, out);
}
